// CausalSelfAttention_OLD_21749714387384
// MI455X (gfx1250) — matrix operand order repaired and checked by reading, not compiled, not run on hardware
//
#include <hip/hip_runtime.h>
#include <stdint.h>

#define B_SZ 2
#define T_SZ 2048
#define C_SZ 2048
#define NH 16
#define NG 4
#define HS 128
#define QKV_O ((NH + 2 * NG) * HS) /* 3072 */
#define SW_SZ 1024
#define ATTN_SCALE 0.08838834764831845f /* 1/sqrt(128) */

typedef __attribute__((ext_vector_type(16))) __bf16 bf16x16;
typedef __attribute__((ext_vector_type(8)))  float  f32x8;
typedef unsigned int u32x4 __attribute__((ext_vector_type(4)));
typedef int          i32x4 __attribute__((ext_vector_type(4)));
typedef int          i32x8 __attribute__((ext_vector_type(8)));

union FragU { bf16x16 v; unsigned u[8]; };

__device__ __forceinline__ f32x8 wmma_bf16(bf16x16 a, bf16x16 b, f32x8 c) {
  // D = A(16x32 bf16) x B(32x16 bf16) + C(16x16 f32)
  return __builtin_amdgcn_wmma_f32_16x16x32_bf16(
      /*neg_a=*/false, a, /*neg_b=*/false, b,
      /*c_mod=*/(short)0, c, /*reuse_a=*/false, /*reuse_b=*/false);
}

__device__ __forceinline__ unsigned short f2bf(float x) {
  unsigned u = __float_as_uint(x);
  u += 0x7FFFu + ((u >> 16) & 1u); // round-to-nearest-even
  return (unsigned short)(u >> 16);
}

// TDM writes LDS behind the compiler's back (the LDS address is carried inside
// an opaque integer descriptor). Escape the buffer address into an asm block
// with a memory clobber so LLVM must assume the buffer is written, otherwise
// the fragment loads from a "never-stored" __shared__ array fold to undef.
__device__ __forceinline__ void dma_visibility_fence(const void* p) {
  asm volatile("" : : "v"(p) : "memory");
}

// A-fragment (16xK=32) from row-major bf16 source; su = row stride in uints.
// Lanes 0-15: row=lane,  K in {0..7, 16..23}; lanes 16-31: row=lane-16, K in {8..15, 24..31}.
__device__ __forceinline__ bf16x16 frag_a(const unsigned* base, int row0, int su) {
  int lane = threadIdx.x & 31;
  int m = row0 + (lane & 15);
  int kbu = (lane < 16) ? 0 : 4; // uint offset of element 0 / 8
  const unsigned* p = base + (size_t)m * su + kbu;
  FragU f;
#pragma unroll
  for (int r = 0; r < 4; ++r) f.u[r] = p[r];
#pragma unroll
  for (int r = 0; r < 4; ++r) f.u[4 + r] = p[8 + r]; // +16 elements
  return f.v;
}

// B-fragment (K=32 x 16) from "n-major" bf16 source Bt[n][k]; su = stride in uints.
// Lanes 0-15: col=lane, K in {0..7, 16..23}; lanes 16-31: col=lane-16, K in {8..15, 24..31}.
__device__ __forceinline__ bf16x16 frag_b(const unsigned* base, int col0, int su) {
  int lane = threadIdx.x & 31;
  int n = col0 + (lane & 15);
  int kbu = (lane < 16) ? 0 : 4; // uint offset of element 0 / 8
  const unsigned* p = base + (size_t)n * su + kbu;
  FragU f;
#pragma unroll
  for (int r = 0; r < 4; ++r) f.u[r] = p[r];
#pragma unroll
  for (int r = 0; r < 4; ++r) f.u[4 + r] = p[8 + r];
  return f.v;
}

// ---------------------------------------------------------------------------
// Tensor Data Mover: async 2D tile load (tile = 32 bf16 x `rows` rows) into LDS.
// D# built per CDNA5 ISA 8.3/8.4: group0 {count, lds_addr, global_addr, type=2},
// group1 {data_size=2B, tensor dims, tile dims, dim0 stride}. Groups 2/3 unused (2D).
// Toolchain builtin is the 6-arg (clang-23) form: (g0, g1, g2, g3, extra, cpol).
// ---------------------------------------------------------------------------
__device__ __forceinline__ void tdm_load_2d(unsigned lds_byte_off, const void* gaddr,
                                            unsigned tensor_rows, unsigned tensor_k,
                                            unsigned rows) {
  unsigned long long ga = (unsigned long long)gaddr;
  u32x4 g0;
  g0[0] = 1u;                                 // count=1, user descriptor
  g0[1] = lds_byte_off;                       // D#.lds_addr (bytes)
  g0[2] = (unsigned)ga;                       // global_addr[31:0]
  g0[3] = (unsigned)(ga >> 32) | 0x80000000u; // global_addr[56:32] | type=2<<30
  i32x8 g1;
  g1[0] = 0x00010000;                                   // data_size=1 (2 bytes)
  g1[1] = (int)((tensor_k & 0xFFFFu) << 16);            // tensor_dim0[15:0]
  g1[2] = (int)(((tensor_k >> 16) & 0xFFFFu) |
                ((tensor_rows & 0xFFFFu) << 16));       // tensor_dim0[31:16] | tensor_dim1[15:0]
  g1[3] = (int)(((tensor_rows >> 16) & 0xFFFFu) |
                (32u << 16));                           // tensor_dim1[31:16] | tile_dim0=32
  g1[4] = (int)rows;                                    // tile_dim1 = rows, tile_dim2 = 0
  g1[5] = (int)tensor_k;                                // tensor_dim0_stride[31:0] = K
  g1[6] = 0;
  g1[7] = 0;
  i32x4 z4 = {0, 0, 0, 0};
  i32x8 z8 = {0, 0, 0, 0, 0, 0, 0, 0};
  __builtin_amdgcn_tensor_load_to_lds(g0, g1, z4, z4, z8, 0);
}

// ---------------------------------------------------------------------------
// Kernel: f32 -> bf16 bits
// ---------------------------------------------------------------------------
__global__ void f32_to_bf16(const float* __restrict__ in,
                            unsigned short* __restrict__ out, int n) {
  int i = blockIdx.x * blockDim.x + threadIdx.x;
  if (i < n) out[i] = f2bf(in[i]);
}

// ---------------------------------------------------------------------------
// Kernel: C[M,N] (f32) = A[M,K] (bf16 row-major) * Bn[N,K]^T (bf16 n-major)
// Block 256 threads (8 waves), tile 128x128, K-step 32.
// Tiles staged via TDM (tensor_load_to_lds), double-buffered, TENSORcnt-paced.
// Per wave: 32x64 register tile = 8 WMMAs per K-step (2 A-frags x 4 B-frags).
// ---------------------------------------------------------------------------
__global__ __launch_bounds__(256) void gemm_bf16(
    const unsigned short* __restrict__ A, const unsigned short* __restrict__ Bn,
    float* __restrict__ C, int M, int N, int K) {
  // [buf0: A(8K) B(8K)][buf1: A(8K) B(8K)] ; rows of 32 bf16 = 64B, row-major
  __shared__ __align__(16) unsigned char smem[4 * 8192];
  int m0 = blockIdx.x * 128;
  int n0 = blockIdx.y * 128;
  int tid = threadIdx.x;
  int w = tid >> 5;
  int waveM = w & 3;  // 4 sub-tiles along M (32 each)
  int waveN = w >> 2; // 2 sub-tiles along N (64 each)
  // Scalar (SGPR) predicate so the TDM issue is a uniform branch, not EXEC-masked.
  bool issuer = (__builtin_amdgcn_readfirstlane(w) == 0);

  // One-time zero fill: gives the staging buffer visible stores so the
  // DMA-filled reads below cannot be reasoned away (also deterministic).
  {
    uint4* sz = (uint4*)smem;
    uint4 zf = {0, 0, 0, 0};
#pragma unroll
    for (int i = 0; i < 8; ++i) sz[tid + i * 256] = zf;
  }
  __syncthreads();

  const f32x8 vz = {0, 0, 0, 0, 0, 0, 0, 0};
  f32x8 acc[2][4];
#pragma unroll
  for (int i = 0; i < 2; ++i)
#pragma unroll
    for (int j = 0; j < 4; ++j) acc[i][j] = vz;

  // Prologue: fill both buffers (wave0 issues; TENSORcnt ops retire in order).
  if (issuer) {
    tdm_load_2d(0,     A  + (size_t)m0 * K, (unsigned)M, (unsigned)K, 128u);
    tdm_load_2d(8192,  Bn + (size_t)n0 * K, (unsigned)N, (unsigned)K, 128u);
    if (K > 32) {
      tdm_load_2d(16384, A  + (size_t)m0 * K + 32, (unsigned)M, (unsigned)K, 128u);
      tdm_load_2d(24576, Bn + (size_t)n0 * K + 32, (unsigned)N, (unsigned)K, 128u);
    }
  }

  int parity = 0;
  for (int k0 = 0; k0 < K; k0 += 32) {
    if (issuer) {
      if (k0 + 32 < K) __builtin_amdgcn_s_wait_tensorcnt(2); // current buf done
      else             __builtin_amdgcn_s_wait_tensorcnt(0); // drain
    }
    __syncthreads();
    dma_visibility_fence(smem); // LDS was DMA-filled: force real ds_loads below
    const unsigned* As_u = (const unsigned*)(smem + parity * 16384);
    const unsigned* Bs_u = (const unsigned*)(smem + parity * 16384 + 8192);
    bf16x16 a0 = frag_a(As_u, waveM * 32, 16);
    bf16x16 a1 = frag_a(As_u, waveM * 32 + 16, 16);
#pragma unroll
    for (int j = 0; j < 4; ++j) {
      bf16x16 bj = frag_b(Bs_u, waveN * 64 + j * 16, 16);
      acc[0][j] = wmma_bf16(a0, bj, acc[0][j]);
      acc[1][j] = wmma_bf16(a1, bj, acc[1][j]);
    }
    __syncthreads(); // everyone done with buf[parity] before it is refilled
    if (issuer && (k0 + 64 < K)) {
      unsigned boff = (unsigned)(parity * 16384);
      tdm_load_2d(boff,        A  + (size_t)m0 * K + (k0 + 64), (unsigned)M, (unsigned)K, 128u);
      tdm_load_2d(boff + 8192, Bn + (size_t)n0 * K + (k0 + 64), (unsigned)N, (unsigned)K, 128u);
    }
    parity ^= 1;
  }

  int lane = tid & 31;
  int col = lane & 15;
  int roff = (lane < 16) ? 0 : 8;
#pragma unroll
  for (int i = 0; i < 2; ++i) {
    int mb = m0 + waveM * 32 + i * 16 + roff;
#pragma unroll
    for (int j = 0; j < 4; ++j) {
      int nb = n0 + waveN * 64 + j * 16;
#pragma unroll
      for (int r = 0; r < 8; ++r)
        C[(size_t)(mb + r) * N + nb + col] = acc[i][j][r];
    }
  }
}

// ---------------------------------------------------------------------------
// Kernel: RoPE on q/k + split into bf16 q[b,h,t,d], k[b,g,t,d], vT[b,g,d,t]
// ---------------------------------------------------------------------------
__global__ void rope_split(const float* __restrict__ qkv,
                           const float* __restrict__ cosT,
                           const float* __restrict__ sinT,
                           unsigned short* __restrict__ qb,
                           unsigned short* __restrict__ kbuf,
                           unsigned short* __restrict__ vtb) {
  int idx = blockIdx.x * blockDim.x + threadIdx.x;
  if (idx >= B_SZ * T_SZ * QKV_O) return;
  int o = idx % QKV_O;
  int bt = idx / QKV_O;
  int t = bt & (T_SZ - 1);
  int b = bt >> 11; // T = 2048
  const float* row = qkv + (size_t)bt * QKV_O;
  if (o < NH * HS) {
    int hh = o >> 7, d = o & 127;
    float x = row[o];
    float p = row[(o & ~127) | ((d < 64) ? (d + 64) : (d - 64))];
    float rot = (d < 64) ? -p : p;
    float v = x * cosT[t * HS + d] + rot * sinT[t * HS + d];
    qb[((size_t)(b * NH + hh) * T_SZ + t) * HS + d] = f2bf(v);
  } else if (o < (NH + NG) * HS) {
    int oo = o - NH * HS;
    int gg = oo >> 7, d = oo & 127;
    float x = row[o];
    float p = row[NH * HS + ((oo & ~127) | ((d < 64) ? (d + 64) : (d - 64)))];
    float rot = (d < 64) ? -p : p;
    float v = x * cosT[t * HS + d] + rot * sinT[t * HS + d];
    kbuf[((size_t)(b * NG + gg) * T_SZ + t) * HS + d] = f2bf(v);
  } else {
    int oo = o - (NH + NG) * HS;
    int gg = oo >> 7, d = oo & 127;
    vtb[((size_t)(b * NG + gg) * HS + d) * T_SZ + t] = f2bf(row[o]); // transpose
  }
}

// ---------------------------------------------------------------------------
// Kernel: flash attention, one 16-row q-tile per wave, 32 keys per iteration.
// Softcap 50*tanh, causal + sliding-window mask, online softmax.
// ---------------------------------------------------------------------------
__global__ __launch_bounds__(128) void attn_fwd(
    const unsigned short* __restrict__ qb, const unsigned short* __restrict__ kbuf,
    const unsigned short* __restrict__ vtb, unsigned short* __restrict__ yb) {
  __shared__ unsigned short Pl[4][16 * 32]; // per-wave P tile (16x32 bf16)
  int lane = threadIdx.x & 31;
  int wid = threadIdx.x >> 5;
  int tile = blockIdx.x * 4 + wid; // 0 .. B*NH*(T/16)-1
  int qt = tile & (T_SZ / 16 - 1);
  int h = (tile >> 7) & (NH - 1);
  int b = tile >> 11;
  int g = h >> 2; // q_per_kv = 4
  int i0 = qt * 16;

  const unsigned* Qu = (const unsigned*)(qb + ((size_t)(b * NH + h) * T_SZ) * HS);
  const unsigned* Ku = (const unsigned*)(kbuf + ((size_t)(b * NG + g) * T_SZ) * HS);
  const unsigned* Vu = (const unsigned*)(vtb + ((size_t)(b * NG + g) * HS) * T_SZ);

  bf16x16 qf[4]; // 16x128 Q tile as four K=32 A-fragments
#pragma unroll
  for (int c = 0; c < 4; ++c)
    qf[c] = frag_a(Qu + (size_t)i0 * 64 + c * 16, 0, 64);

  float mrow[8], lrow[8];
  f32x8 oacc[8]; // 16x128 output accumulator
  const f32x8 vzero = {0, 0, 0, 0, 0, 0, 0, 0};
#pragma unroll
  for (int r = 0; r < 8; ++r) { mrow[r] = -1e30f; lrow[r] = 0.f; }
#pragma unroll
  for (int o = 0; o < 8; ++o) oacc[o] = vzero;

  int col = lane & 15;
  int roff = (lane < 16) ? 0 : 8;
  int jlo = i0 - SW_SZ + 1;
  if (jlo < 0) jlo = 0;
  int jstart = jlo & ~31;
  int jend = i0 + 15;

  for (int t0 = jstart; t0 <= jend; t0 += 32) {
    f32x8 s0 = vzero, s1 = vzero; // scores vs keys [t0,t0+16) and [t0+16,t0+32)
#pragma unroll
    for (int c = 0; c < 4; ++c) {
      bf16x16 kf0 = frag_b(Ku + (size_t)t0 * 64 + c * 16, 0, 64);
      bf16x16 kf1 = frag_b(Ku + (size_t)(t0 + 16) * 64 + c * 16, 0, 64);
      s0 = wmma_bf16(qf[c], kf0, s0);
      s1 = wmma_bf16(qf[c], kf1, s1);
    }
#pragma unroll
    for (int r = 0; r < 8; ++r) {
      int i = i0 + r + roff;
      float a0 = 50.0f * tanhf(s0[r] * ATTN_SCALE * (1.0f / 50.0f));
      float a1 = 50.0f * tanhf(s1[r] * ATTN_SCALE * (1.0f / 50.0f));
      int j0 = t0 + col, j1 = t0 + 16 + col;
      bool ok0 = (j0 <= i) && (j0 > i - SW_SZ);
      bool ok1 = (j1 <= i) && (j1 > i - SW_SZ);
      a0 = ok0 ? a0 : -1e30f;
      a1 = ok1 ? a1 : -1e30f;
      float rm = fmaxf(a0, a1); // row max across the 16-lane half
#pragma unroll
      for (int sh = 1; sh < 16; sh <<= 1) rm = fmaxf(rm, __shfl_xor(rm, sh, 32));
      float mnew = fmaxf(mrow[r], rm);
      float alpha = __expf(mrow[r] - mnew);
      float p0 = ok0 ? __expf(a0 - mnew) : 0.f;
      float p1 = ok1 ? __expf(a1 - mnew) : 0.f;
      float ps = p0 + p1;
#pragma unroll
      for (int sh = 1; sh < 16; sh <<= 1) ps += __shfl_xor(ps, sh, 32);
      lrow[r] = lrow[r] * alpha + ps;
      mrow[r] = mnew;
#pragma unroll
      for (int o = 0; o < 8; ++o) oacc[o][r] *= alpha;
      Pl[wid][(r + roff) * 32 + col] = f2bf(p0);        // relayout C->A via LDS
      Pl[wid][(r + roff) * 32 + col + 16] = f2bf(p1);
    }
    asm volatile("s_wait_dscnt 0" ::: "memory"); // wave-local LDS ordering
    bf16x16 pf = frag_a((const unsigned*)&Pl[wid][0], 0, 16);
#pragma unroll
    for (int o = 0; o < 8; ++o) {
      bf16x16 vf = frag_b(Vu + (t0 >> 1), o * 16, 1024); // V^T: d-major, t contiguous
      oacc[o] = wmma_bf16(pf, vf, oacc[o]);
    }
    asm volatile("s_wait_dscnt 0" ::: "memory");
  }

#pragma unroll
  for (int r = 0; r < 8; ++r) {
    float inv = 1.0f / lrow[r];
    int t = i0 + r + roff;
#pragma unroll
    for (int o = 0; o < 8; ++o) {
      int d = o * 16 + col;
      yb[((size_t)b * T_SZ + t) * (NH * HS) + h * HS + d] = f2bf(oacc[o][r] * inv);
    }
  }
}

// ---------------------------------------------------------------------------
extern "C" void kernel_launch(void* const* d_in, const int* in_sizes, int n_in,
                              void* d_out, int out_size, void* d_ws, size_t ws_size,
                              hipStream_t stream) {
  (void)in_sizes; (void)n_in; (void)out_size; (void)ws_size;
  const float* x     = (const float*)d_in[0];
  const float* cosT  = (const float*)d_in[1];
  const float* sinT  = (const float*)d_in[2];
  const float* Wqkv  = (const float*)d_in[3];
  const float* Wproj = (const float*)d_in[4];
  float* out = (float*)d_out;

  char* ws = (char*)d_ws;
  size_t off = 0;
  auto take = [&](size_t bytes) {
    char* p = ws + off;
    off = (off + bytes + 255) & ~(size_t)255;
    return p;
  };
  float*          qkv_f    = (float*)take((size_t)B_SZ * T_SZ * QKV_O * 4);
  unsigned short* x_bf     = (unsigned short*)take((size_t)B_SZ * T_SZ * C_SZ * 2);
  unsigned short* wqkv_bf  = (unsigned short*)take((size_t)QKV_O * C_SZ * 2);
  unsigned short* wproj_bf = (unsigned short*)take((size_t)C_SZ * NH * HS * 2);
  unsigned short* q_bf     = (unsigned short*)take((size_t)B_SZ * NH * T_SZ * HS * 2);
  unsigned short* k_bf     = (unsigned short*)take((size_t)B_SZ * NG * T_SZ * HS * 2);
  unsigned short* vt_bf    = (unsigned short*)take((size_t)B_SZ * NG * T_SZ * HS * 2);
  unsigned short* y_bf     = (unsigned short*)take((size_t)B_SZ * T_SZ * NH * HS * 2);

  int nx = B_SZ * T_SZ * C_SZ;
  int nq = QKV_O * C_SZ;
  int np = C_SZ * NH * HS;
  f32_to_bf16<<<(nx + 255) / 256, 256, 0, stream>>>(x, x_bf, nx);
  f32_to_bf16<<<(nq + 255) / 256, 256, 0, stream>>>(Wqkv, wqkv_bf, nq);
  f32_to_bf16<<<(np + 255) / 256, 256, 0, stream>>>(Wproj, wproj_bf, np);

  // qkv[BT,3072] = x[BT,2048] @ Wqkv^T   (128x128 tiles)
  gemm_bf16<<<dim3((B_SZ * T_SZ) / 128, QKV_O / 128), 256, 0, stream>>>(
      x_bf, wqkv_bf, qkv_f, B_SZ * T_SZ, QKV_O, C_SZ);

  int nr = B_SZ * T_SZ * QKV_O;
  rope_split<<<(nr + 255) / 256, 256, 0, stream>>>(qkv_f, cosT, sinT, q_bf, k_bf, vt_bf);

  // 2*16*128 q-tiles, 4 per block
  attn_fwd<<<(B_SZ * NH * (T_SZ / 16)) / 4, 128, 0, stream>>>(q_bf, k_bf, vt_bf, y_bf);

  // out[BT,2048] = y[BT,2048] @ Wproj^T
  gemm_bf16<<<dim3((B_SZ * T_SZ) / 128, C_SZ / 128), 256, 0, stream>>>(
      y_bf, wproj_bf, out, B_SZ * T_SZ, C_SZ, C_SZ);
}
